// StatefulMoPPolicy_52338471469236
// MI455X (gfx1250) — matrix operand order repaired and checked by reading, not compiled, not run on hardware
//
#include <hip/hip_runtime.h>
#include <math.h>

// ---------------------------------------------------------------------------
// StatefulMoPPolicy for MI455X (gfx1250).
// Heavy GEMMs -> bf16 WMMA (v_wmma_f32_16x16x32_bf16), f32 accumulate.
// 128x128 block tile, 8 waves, 32x64 per wave = 8 WMMA per K-step.
// Tile fill via GLOBAL_LOAD_ASYNC_TO_LDS_B128 (ASYNCcnt) -> no staging VGPRs,
// no scratch spill; async copy of tile k+1 overlaps WMMA burst on tile k.
// ---------------------------------------------------------------------------

#define B_ROWS 8192
#define EPSF   1e-5f

typedef __attribute__((ext_vector_type(16))) __bf16 v16bf;
typedef __attribute__((ext_vector_type(8)))  float  v8f;

struct alignas(16) Q4 { unsigned int x[4]; };
union UFrag { v16bf v; Q4 q[2]; };

__device__ __forceinline__ unsigned short f2bf(float f) {
  union { float f; unsigned int u; } v; v.f = f;
  unsigned int r = v.u + 0x7FFFu + ((v.u >> 16) & 1u);   // round-to-nearest-even
  return (unsigned short)(r >> 16);
}
__device__ __forceinline__ unsigned int pack2bf(float a, float b) {
  return (unsigned int)f2bf(a) | ((unsigned int)f2bf(b) << 16);
}
__device__ __forceinline__ float sigmoidf_(float x) { return 1.f / (1.f + expf(-x)); }

// ---------------------------------------------------------------------------
// GEMM: C[M,N] = A[M,K] @ W[N,K]^T + bias[N]     (A, W already bf16)
// K % 32 == 0, M % 128 == 0, N % 128 == 0 (true for all shapes here).
// ---------------------------------------------------------------------------
#define BM 128
#define BN 128
#define BK 32
#define LDSW 40   // halfs per LDS row = 80 bytes (multiple of 16B)

// async copy 32 bytes: global (64-bit vaddr) -> LDS (32-bit lds offset).
// INST_OFFSET is added to BOTH the LDS and global addresses (ISA 10.4).
__device__ __forceinline__ void async_copy32(unsigned lds_off,
                                             const unsigned short* gptr) {
  asm volatile("global_load_async_to_lds_b128 %0, %1, off"
               :: "v"(lds_off), "v"(gptr) : "memory");
  asm volatile("global_load_async_to_lds_b128 %0, %1, off offset:16"
               :: "v"(lds_off), "v"(gptr) : "memory");
}
__device__ __forceinline__ void wait_async0() {
  asm volatile("s_wait_asynccnt 0x0" ::: "memory");
}

__global__ __launch_bounds__(256, 2) void gemm_bias_wmma_bf16(
    const unsigned short* __restrict__ A, const unsigned short* __restrict__ W,
    const float* __restrict__ bias, float* __restrict__ C,
    int M, int N, int K)
{
  __shared__ __align__(16) unsigned short As[2][BM][LDSW];
  __shared__ __align__(16) unsigned short Ws[2][BN][LDSW];

  const int tid  = threadIdx.x;
  const int lane = tid & 31;
  const int wave = tid >> 5;
  const int wm   = wave & 3;     // wave row (M): 0..3, 32 rows each
  const int wn   = wave >> 2;    // wave col (N): 0..1, 64 cols each
  const int m0   = blockIdx.y * BM;
  const int n0   = blockIdx.x * BN;

  // tile loader mapping: 128 rows x 32 halfs, 16 halfs (32B) per thread
  const int lr  = tid >> 1;          // 0..127
  const int lch = (tid & 1) * 16;    // 0 or 16 (halfs)

  // LDS byte offsets for this thread's async destinations (both buffers)
  const unsigned ldsA[2] = { (unsigned)(size_t)&As[0][lr][lch],
                             (unsigned)(size_t)&As[1][lr][lch] };
  const unsigned ldsW[2] = { (unsigned)(size_t)&Ws[0][lr][lch],
                             (unsigned)(size_t)&Ws[1][lr][lch] };
  const unsigned short* gA = A + (size_t)(m0 + lr) * K + lch;
  const unsigned short* gW = W + (size_t)(n0 + lr) * K + lch;

  v8f acc[2][4];
#pragma unroll
  for (int a = 0; a < 2; ++a)
#pragma unroll
    for (int b = 0; b < 4; ++b)
#pragma unroll
      for (int i = 0; i < 8; ++i) acc[a][b][i] = 0.f;

  // fragment addressing (ISA 7.12.2, wave32 16-bit layouts)
  const int akb = (lane < 16) ? 0 : 8;    // A: K pairs 0..7 / 8..15 (+16 hi VGPRs)
  const int bkb = (lane < 16) ? 0 : 8;    // B: K 0..7 / 8..15 (+16 hi VGPRs)
  const int l15 = lane & 15;

  // prologue: async tile 0 -> buffer 0
  async_copy32(ldsA[0], gA);
  async_copy32(ldsW[0], gW);

  int buf = 0;
  for (int k0 = 0; k0 < K; k0 += BK, buf ^= 1) {
    wait_async0();     // this wave's tile-k copies landed in LDS
    __syncthreads();   // everyone's tile-k copies visible; buf^1 reads done

    if (k0 + BK < K) { // async copy of tile k+1 overlaps the WMMA burst
      const int nb = buf ^ 1;
      async_copy32(ldsA[nb], gA + (k0 + BK));
      async_copy32(ldsW[nb], gW + (k0 + BK));
      if (k0 + 2 * BK < K) {   // warm L2 two tiles ahead
        __builtin_prefetch(gA + (k0 + 2 * BK), 0, 1);
        __builtin_prefetch(gW + (k0 + 2 * BK), 0, 1);
      }
    }

    UFrag ua[2], ub[4];
#pragma unroll
    for (int am = 0; am < 2; ++am) {
      const int r = wm * 32 + am * 16 + l15;
      ua[am].q[0] = *reinterpret_cast<const Q4*>(&As[buf][r][akb]);
      ua[am].q[1] = *reinterpret_cast<const Q4*>(&As[buf][r][akb + 16]);
    }
#pragma unroll
    for (int bn = 0; bn < 4; ++bn) {
      const int r = wn * 64 + bn * 16 + l15;
      ub[bn].q[0] = *reinterpret_cast<const Q4*>(&Ws[buf][r][bkb]);
      ub[bn].q[1] = *reinterpret_cast<const Q4*>(&Ws[buf][r][bkb + 16]);
    }

#pragma unroll
    for (int am = 0; am < 2; ++am)
#pragma unroll
      for (int bn = 0; bn < 4; ++bn)
        acc[am][bn] = __builtin_amdgcn_wmma_f32_16x16x32_bf16(
            false, ua[am].v, false, ub[bn].v, (short)0, acc[am][bn], false, false);
  }

  // C/D layout: VGPR r -> row r (+8 for lanes 16..31), lane&15 -> column
  const int mrow = m0 + wm * 32 + ((lane >> 4) << 3);
  const int ncol = n0 + wn * 64 + l15;
#pragma unroll
  for (int bn = 0; bn < 4; ++bn) {
    const int n = ncol + bn * 16;
    const float bv = bias[n];
#pragma unroll
    for (int am = 0; am < 2; ++am) {
      const int m = mrow + am * 16;
#pragma unroll
      for (int r = 0; r < 8; ++r)
        C[(size_t)(m + r) * N + n] = acc[am][bn][r] + bv;
    }
  }
}

// ---------------------------------------------------------------------------
// Elementwise / reduction kernels
// ---------------------------------------------------------------------------

// packed f32 -> bf16 (8 elements / thread); n % 8 == 0
__global__ __launch_bounds__(256) void f32_to_bf16_kernel(
    const float* __restrict__ in, unsigned short* __restrict__ out, int n)
{
  int base = (blockIdx.x * 256 + threadIdx.x) * 8;
  if (base >= n) return;
  const float4* p = reinterpret_cast<const float4*>(in + base);
  float4 a = p[0], b = p[1];
  Q4 q;
  q.x[0] = pack2bf(a.x, a.y);
  q.x[1] = pack2bf(a.z, a.w);
  q.x[2] = pack2bf(b.x, b.y);
  q.x[3] = pack2bf(b.z, b.w);
  *reinterpret_cast<Q4*>(out + base) = q;
}

// xc_bf16 = concat(x, lang)
__global__ __launch_bounds__(256) void concat_bf16_kernel(
    const float* __restrict__ x, const float* __restrict__ l,
    unsigned short* __restrict__ xc, int n)
{
  int i = blockIdx.x * 256 + threadIdx.x;
  if (i >= n) return;
  int m = i >> 9;          // /512
  int c = i & 511;
  float v = (c < 64) ? x[m * 64 + c] : l[m * 448 + (c - 64)];
  xc[i] = f2bf(v);
}

__global__ __launch_bounds__(256) void gru_kernel(
    const float* __restrict__ gi, const float* __restrict__ gh,
    const float* __restrict__ h, float* __restrict__ hnew, int n, int H)
{
  int i = blockIdx.x * 256 + threadIdx.x;
  if (i >= n) return;
  int m = i / H, j = i - m * H;
  const float* gir = gi + (size_t)m * 3 * H;
  const float* ghr = gh + (size_t)m * 3 * H;
  float r  = sigmoidf_(gir[j]     + ghr[j]);
  float z  = sigmoidf_(gir[H + j] + ghr[H + j]);
  float nn = tanhf(gir[2 * H + j] + r * ghr[2 * H + j]);
  hnew[i] = (1.f - z) * nn + z * h[i];
}

// gate = softmax(relu(h_r) @ r_Wo^T + r_bo), RD=256, NE=4. Wave per row.
__global__ __launch_bounds__(256) void gate_kernel(
    const float* __restrict__ h, const float* __restrict__ Wo,
    const float* __restrict__ bo, float* __restrict__ gate, int Bn)
{
  int row  = blockIdx.x * 8 + (threadIdx.x >> 5);
  int lane = threadIdx.x & 31;
  if (row >= Bn) return;
  float acc[4] = {0.f, 0.f, 0.f, 0.f};
  for (int t = 0; t < 8; ++t) {
    int k = lane + t * 32;
    float hv = fmaxf(h[(size_t)row * 256 + k], 0.f);
#pragma unroll
    for (int e = 0; e < 4; ++e) acc[e] += hv * Wo[e * 256 + k];
  }
#pragma unroll
  for (int e = 0; e < 4; ++e)
    for (int s = 16; s > 0; s >>= 1) acc[e] += __shfl_xor(acc[e], s, 32);
#pragma unroll
  for (int e = 0; e < 4; ++e) acc[e] += bo[e];
  float mx = fmaxf(fmaxf(acc[0], acc[1]), fmaxf(acc[2], acc[3]));
  float ex[4], ssum = 0.f;
#pragma unroll
  for (int e = 0; e < 4; ++e) { ex[e] = expf(acc[e] - mx); ssum += ex[e]; }
  if (lane < 4) gate[(size_t)row * 4 + lane] = ex[lane] / ssum;
}

__global__ __launch_bounds__(256) void final_init_kernel(
    const float* __restrict__ gate, const float* __restrict__ x,
    float* __restrict__ fin, int n)
{
  int i = blockIdx.x * 256 + threadIdx.x;
  if (i >= n) return;
  int m = i >> 10;   // /1024
  fin[i] = gate[m * 4 + 3] * x[i];
}

__global__ __launch_bounds__(256) void final_accum_kernel(
    const float* __restrict__ gate, const float* __restrict__ y,
    float* __restrict__ fin, int n, int e)
{
  int i = blockIdx.x * 256 + threadIdx.x;
  if (i >= n) return;
  int m = i >> 10;
  fin[i] += gate[m * 4 + e] * y[i];
}

// per-feature batch statistics over 8192 rows (H=512), coalesced columns
__global__ __launch_bounds__(256) void bn_stats_kernel(
    const float* __restrict__ h, float* __restrict__ mu,
    float* __restrict__ rstd, int Bn, int H)
{
  int col = blockIdx.x * 32 + (threadIdx.x & 31);
  int ty  = threadIdx.x >> 5;
  float s = 0.f, s2 = 0.f;
  for (int r = ty; r < Bn; r += 8) {
    float v = h[(size_t)r * H + col];
    s += v; s2 += v * v;
  }
  __shared__ float sh1[8][32], sh2[8][32];
  sh1[ty][threadIdx.x & 31] = s;
  sh2[ty][threadIdx.x & 31] = s2;
  __syncthreads();
  if (ty == 0) {
    for (int w = 1; w < 8; ++w) { s += sh1[w][threadIdx.x & 31]; s2 += sh2[w][threadIdx.x & 31]; }
    float mean = s / (float)Bn;
    float var  = s2 / (float)Bn - mean * mean;
    mu[col]   = mean;
    rstd[col] = rsqrtf(var + EPSF);
  }
}

// y_bf16 = relu(bn(h))  -- feeds the Wo GEMM directly
__global__ __launch_bounds__(256) void bn_relu_bf16_kernel(
    const float* __restrict__ h, const float* __restrict__ mu,
    const float* __restrict__ rstd, const float* __restrict__ g,
    const float* __restrict__ b, unsigned short* __restrict__ y, int n, int H)
{
  int i = blockIdx.x * 256 + threadIdx.x;
  if (i >= n) return;
  int j = i % H;
  y[i] = f2bf(fmaxf((h[i] - mu[j]) * rstd[j] * g[j] + b[j], 0.f));
}

// out = layernorm(x + fin) * g + b over D=1024; writes fp32 + bf16. Wave/row.
__global__ __launch_bounds__(256) void add_ln_kernel(
    const float* __restrict__ x, const float* __restrict__ fin,
    const float* __restrict__ g, const float* __restrict__ b,
    float* __restrict__ out, unsigned short* __restrict__ out_bf, int Bn)
{
  int row  = blockIdx.x * 8 + (threadIdx.x >> 5);
  int lane = threadIdx.x & 31;
  if (row >= Bn) return;
  float v[32];
  float s = 0.f, s2 = 0.f;
#pragma unroll
  for (int t = 0; t < 32; ++t) {
    int j = lane + t * 32;
    float val = x[(size_t)row * 1024 + j] + fin[(size_t)row * 1024 + j];
    v[t] = val; s += val; s2 += val * val;
  }
  for (int m = 16; m > 0; m >>= 1) {
    s  += __shfl_xor(s,  m, 32);
    s2 += __shfl_xor(s2, m, 32);
  }
  float mean = s * (1.f / 1024.f);
  float var  = s2 * (1.f / 1024.f) - mean * mean;
  float rstd = rsqrtf(var + EPSF);
#pragma unroll
  for (int t = 0; t < 32; ++t) {
    int j = lane + t * 32;
    float r = (v[t] - mean) * rstd * g[j] + b[j];
    out[(size_t)row * 1024 + j] = r;
    out_bf[(size_t)row * 1024 + j] = f2bf(r);
  }
}

// out[B,18] = x[B,1024] @ W_out[18,1024]^T + b_out. Wave per row.
__global__ __launch_bounds__(256) void out_kernel(
    const float* __restrict__ x, const float* __restrict__ Wout,
    const float* __restrict__ bout, float* __restrict__ out, int Bn)
{
  int row  = blockIdx.x * 8 + (threadIdx.x >> 5);
  int lane = threadIdx.x & 31;
  if (row >= Bn) return;
  float acc[18];
#pragma unroll
  for (int e = 0; e < 18; ++e) acc[e] = 0.f;
  for (int t = 0; t < 32; ++t) {
    int k = lane + t * 32;
    float xv = x[(size_t)row * 1024 + k];
#pragma unroll
    for (int e = 0; e < 18; ++e) acc[e] += xv * Wout[e * 1024 + k];
  }
#pragma unroll
  for (int e = 0; e < 18; ++e)
    for (int s = 16; s > 0; s >>= 1) acc[e] += __shfl_xor(acc[e], s, 32);
  if (lane < 18) out[(size_t)row * 18 + lane] = acc[lane] + bout[lane];
}

// ---------------------------------------------------------------------------
// Host orchestration
// ---------------------------------------------------------------------------
static inline void gemm(const unsigned short* A, const unsigned short* W,
                        const float* bias, float* C, int M, int N, int K,
                        hipStream_t s) {
  dim3 grid(N / BN, M / BM);
  gemm_bias_wmma_bf16<<<grid, 256, 0, s>>>(A, W, bias, C, M, N, K);
}
static inline void cvt(const float* in, unsigned short* out, int n, hipStream_t s) {
  f32_to_bf16_kernel<<<(n / 8 + 255) / 256, 256, 0, s>>>(in, out, n);
}

extern "C" void kernel_launch(void* const* d_in, const int* in_sizes, int n_in,
                              void* d_out, int out_size, void* d_ws, size_t ws_size,
                              hipStream_t stream) {
  (void)in_sizes; (void)n_in; (void)out_size; (void)ws_size;
  const int B = B_ROWS;

  const float* x    = (const float*)d_in[0];
  const float* lang = (const float*)d_in[1];
  const float* h_r[2]    = { (const float*)d_in[2], (const float*)d_in[3] };
  const float* h_e[2][3] = {
    { (const float*)d_in[4], (const float*)d_in[5], (const float*)d_in[6] },
    { (const float*)d_in[7], (const float*)d_in[8], (const float*)d_in[9] } };
  const float* W_in  = (const float*)d_in[10];
  const float* b_in  = (const float*)d_in[11];
  const float* W_out = (const float*)d_in[76];
  const float* b_out = (const float*)d_in[77];

  // ---- workspace layout: fp32 region, then bf16 region (16B aligned) ----
  float* ws = (float*)d_ws;
  size_t o = 0;
  float* xpA  = ws + o; o += (size_t)B * 1024;
  float* xpB  = ws + o; o += (size_t)B * 1024;
  float* gi   = ws + o; o += (size_t)B * 1536;
  float* gh   = ws + o; o += (size_t)B * 1536;
  float* hrn  = ws + o; o += (size_t)B * 256;
  float* gate = ws + o; o += (size_t)B * 4;
  float* hn   = ws + o; o += (size_t)B * 512;
  float* fin  = ws + o; o += (size_t)B * 1024;
  float* mu   = ws + o; o += 512;
  float* rstd = ws + o; o += 512;
  o = (o + 63) & ~(size_t)63;
  unsigned short* us = (unsigned short*)(ws + o);
  size_t u = 0;
  unsigned short* xc_bf   = us + u; u += (size_t)B * 512;
  unsigned short* xp_bf   = us + u; u += (size_t)B * 1024;
  unsigned short* h_bf    = us + u; u += (size_t)B * 512;
  unsigned short* yact_bf = us + u; u += (size_t)B * 512;
  unsigned short* wbuf    = us + u; u += (size_t)1536 * 1024;
  float* y = gi;   // gi dead once GRU gates consumed -> reuse for expert output

  const int nD = B * 1024;
  const int nE = B * 512;
  const int gD = (nD + 255) / 256;
  const int gE = (nE + 255) / 256;
  const int gW = B / 8;   // wave-per-row kernels

  // xp = concat(x, lang) @ W_in^T + b_in
  concat_bf16_kernel<<<gE, 256, 0, stream>>>(x, lang, xc_bf, nE);
  cvt(W_in, wbuf, 1024 * 512, stream);
  gemm(xc_bf, wbuf, b_in, xpA, B, 1024, 512, stream);
  cvt(xpA, xp_bf, nD, stream);

  float* xp_cur = xpA;
  float* xp_nxt = xpB;

  for (int bi = 0; bi < 2; ++bi) {
    const int pb = 12 + bi * 32;
    const float* r_Wih = (const float*)d_in[pb + 0];
    const float* r_Whh = (const float*)d_in[pb + 1];
    const float* r_bih = (const float*)d_in[pb + 2];
    const float* r_bhh = (const float*)d_in[pb + 3];
    const float* r_Wo  = (const float*)d_in[pb + 4];
    const float* r_bo  = (const float*)d_in[pb + 5];
    const float* ln_g  = (const float*)d_in[pb + 6];
    const float* ln_b  = (const float*)d_in[pb + 7];

    // router GRU (RD = 256)
    cvt(r_Wih, wbuf, 768 * 1024, stream);
    gemm(xp_bf, wbuf, r_bih, gi, B, 768, 1024, stream);
    cvt(h_r[bi], h_bf, B * 256, stream);
    cvt(r_Whh, wbuf, 768 * 256, stream);
    gemm(h_bf, wbuf, r_bhh, gh, B, 768, 256, stream);
    gru_kernel<<<(B * 256 + 255) / 256, 256, 0, stream>>>(gi, gh, h_r[bi], hrn, B * 256, 256);
    gate_kernel<<<gW, 256, 0, stream>>>(hrn, r_Wo, r_bo, gate, B);
    final_init_kernel<<<gD, 256, 0, stream>>>(gate, xp_cur, fin, nD);

    for (int e = 0; e < 3; ++e) {
      const int eb = pb + 8 + e * 8;
      const float* Wih  = (const float*)d_in[eb + 0];
      const float* Whh  = (const float*)d_in[eb + 1];
      const float* bih  = (const float*)d_in[eb + 2];
      const float* bhh  = (const float*)d_in[eb + 3];
      const float* bn_g = (const float*)d_in[eb + 4];
      const float* bn_b = (const float*)d_in[eb + 5];
      const float* Wo   = (const float*)d_in[eb + 6];
      const float* bo   = (const float*)d_in[eb + 7];

      // expert GRU (ED = 512)
      cvt(Wih, wbuf, 1536 * 1024, stream);
      gemm(xp_bf, wbuf, bih, gi, B, 1536, 1024, stream);
      cvt(h_e[bi][e], h_bf, B * 512, stream);
      cvt(Whh, wbuf, 1536 * 512, stream);
      gemm(h_bf, wbuf, bhh, gh, B, 1536, 512, stream);
      gru_kernel<<<gE, 256, 0, stream>>>(gi, gh, h_e[bi][e], hn, nE, 512);

      // batchnorm over batch dim + relu (bf16 out, feeds Wo GEMM)
      bn_stats_kernel<<<512 / 32, 256, 0, stream>>>(hn, mu, rstd, B, 512);
      bn_relu_bf16_kernel<<<gE, 256, 0, stream>>>(hn, mu, rstd, bn_g, bn_b, yact_bf, nE, 512);

      // y = relu_bn @ Wo^T + bo ; fin += gate[:,e] * y
      cvt(Wo, wbuf, 1024 * 512, stream);
      gemm(yact_bf, wbuf, bo, y, B, 1024, 512, stream);
      final_accum_kernel<<<gD, 256, 0, stream>>>(gate, y, fin, nD, e);
    }

    // xp_next = layernorm(xp + fin) * ln_g + ln_b   (fp32 + bf16 for next GEMMs)
    add_ln_kernel<<<gW, 256, 0, stream>>>(xp_cur, fin, ln_g, ln_b, xp_nxt, xp_bf, B);
    float* t = xp_cur; xp_cur = xp_nxt; xp_nxt = t;
  }

  out_kernel<<<gW, 256, 0, stream>>>(xp_cur, W_out, b_out, (float*)d_out, B);
}
